// RNNSeq2SeqEncoder_12189117186696
// MI455X (gfx1250) — matrix operand order repaired and checked by reading, not compiled, not run on hardware
//
#include <hip/hip_runtime.h>
#include <hip/hip_bf16.h>

// ---------------------------------------------------------------------------
// CDNA5 / gfx1250 bidirectional 2-layer LSTM encoder.
// Matrix math on v_wmma_f32_16x16x32_bf16 (f32 accumulate); A tiles staged
// through LDS with global_load_async_to_lds_b128 (ASYNCcnt) in the GEMMs.
// ---------------------------------------------------------------------------

typedef __bf16 bf16_t;
typedef __attribute__((ext_vector_type(16))) __bf16 v16bf;
typedef __attribute__((ext_vector_type(8)))  __bf16 v8bf;
typedef __attribute__((ext_vector_type(8)))  float  v8f;

#define NLAYERS 2
#define Bn 8
#define Tn 2048
#define Dn 512
#define Hn 512
#define G4 (4 * Hn)   // 2048 gate columns
#define H2 (2 * Hn)   // 1024 concat width
#define MT (Bn * Tn)  // 16384 rows for the big GEMMs

__device__ __forceinline__ float sigmoidf_(float x) {
  return 1.0f / (1.0f + __expf(-x));
}

// Low 32 bits of the generic address of a __shared__ object = LDS byte addr.
__device__ __forceinline__ uint32_t lds_addr_of(const void* p) {
  return (uint32_t)(uintptr_t)p;
}

// Async 16B copy global -> LDS (per active lane). Tracked by ASYNCcnt.
__device__ __forceinline__ void async_copy_b128(uint32_t lds_off,
                                                const bf16_t* g) {
  asm volatile("global_load_async_to_lds_b128 %0, %1, off"
               :
               : "v"(lds_off), "v"(g)
               : "memory");
}
__device__ __forceinline__ void wait_async0() {
  asm volatile("s_wait_asynccnt 0x0" ::: "memory");
}

// --------------------------------------------------------------------------
// f32 -> bf16 elementwise convert
// --------------------------------------------------------------------------
__global__ void f32_to_bf16_kernel(const float* __restrict__ src,
                                   bf16_t* __restrict__ dst, int n) {
  int i = blockIdx.x * blockDim.x + threadIdx.x;
  if (i < n) dst[i] = (bf16_t)src[i];
}

// --------------------------------------------------------------------------
// Repack a K x N row-major f32 weight into WMMA B-operand fragment-major
// bf16 layout. Fragment (nt, kt) holds a 32(K) x 16(N) tile; within it,
// lane L supplies column (L & 15), K values 16*(L>>4) + j  (contiguous k
// order, same as the A loads). Each fragment is 512 contiguous bf16 -> each lane's
// hot-loop load is one contiguous 32B read.
// --------------------------------------------------------------------------
__global__ void repack_b_kernel(const float* __restrict__ src,
                                bf16_t* __restrict__ dst, int K, int N) {
  int o = blockIdx.x * blockDim.x + threadIdx.x;
  if (o >= K * N) return;
  int frag = o >> 9;
  int rem  = o & 511;
  int lane = rem >> 4;
  int j    = rem & 15;
  int ktiles = K >> 5;
  int nt = frag / ktiles;
  int kt = frag - nt * ktiles;
  int n = nt * 16 + (lane & 15);
  int k = kt * 32 + ((lane >> 4) << 4) + j;
  dst[o] = (bf16_t)src[(size_t)k * N + n];
}

// --------------------------------------------------------------------------
// flip_sequences on a bf16 (B,T,D) tensor
// --------------------------------------------------------------------------
__global__ void flip_bf16_kernel(const bf16_t* __restrict__ x,
                                 bf16_t* __restrict__ xr,
                                 const int* __restrict__ seq, int n) {
  int i = blockIdx.x * blockDim.x + threadIdx.x;
  if (i >= n) return;
  int d  = i % Dn;
  int bt = i / Dn;
  int t  = bt % Tn;
  int b  = bt / Tn;
  int L  = seq[b];
  int ts = (t < L) ? (L - 1 - t) : t;
  xr[i] = x[((size_t)b * Tn + ts) * Dn + d];
}

// --------------------------------------------------------------------------
// WMMA GEMM: C[M,N] = A[M,K] (bf16 row-major) * Bp (packed bf16) + bias.
// A block (8 waves) shares one 16-row M tile and covers 512 N columns;
// the 16x32 A tile per K-step is staged in LDS via async-to-LDS copies
// (double buffered, issued by waves 0-1, published with s_wait_asynccnt +
// barrier) so A global traffic is shared 8 ways. Each wave computes a
// 16(M) x 64(N) tile (4 WMMA accumulators); B fragments are contiguous
// 32B loads that stay L2-resident.
// --------------------------------------------------------------------------
__global__ __launch_bounds__(256) void wmma_gemm_kernel(
    const bf16_t* __restrict__ A, const bf16_t* __restrict__ Bp,
    const float* __restrict__ bias, float* __restrict__ Cf,
    bf16_t* __restrict__ Cbf, int M, int N, int K) {
  __shared__ __align__(16) bf16_t a_sh[2][16 * 32];  // 2 x 1KB

  const int tid  = threadIdx.x;
  const int lane = tid & 31;
  const int wv   = tid >> 5;  // 0..7
  const int nsuper = N >> 9;  // 512-column supergroups per M tile
  const int mt = blockIdx.x / nsuper;
  const int ng = (blockIdx.x - mt * nsuper) * 8 + wv;  // 64-col group
  const int ktiles = K >> 5;
  const int khalf  = (lane >> 4) * 16;
  const int ntbase = ng * 4;

  // A staging: threads 0..63 each copy one 16B segment per K-step.
  const bf16_t* ag = A + ((size_t)mt * 16 + (tid >> 2)) * K + (tid & 3) * 8;
  uint32_t ldsA[2];
  ldsA[0] = lds_addr_of(&a_sh[0][(tid >> 2) * 32 + (tid & 3) * 8]);
  ldsA[1] = lds_addr_of(&a_sh[1][(tid >> 2) * 32 + (tid & 3) * 8]);
  if (tid < 64) async_copy_b128(ldsA[0], ag);  // kt = 0

  v8f acc[4];
  for (int g = 0; g < 4; ++g)
    for (int r = 0; r < 8; ++r) acc[g][r] = 0.0f;

  for (int kt = 0; kt < ktiles; ++kt) {
    const int buf = kt & 1;
    if (tid < 64) wait_async0();  // waves 0-1: fill of a_sh[buf] done
    __syncthreads();              // publish tile (also: prior reads done)
    if (tid < 64 && kt + 1 < ktiles)
      async_copy_b128(ldsA[buf ^ 1], ag + (kt + 1) * 32);

    const bf16_t* ap = &a_sh[buf][(lane & 15) * 32 + khalf];
    v8bf a0 = *(const v8bf*)(ap);
    v8bf a1 = *(const v8bf*)(ap + 8);
    v16bf a;
    for (int j = 0; j < 8; ++j) { a[j] = a0[j]; a[8 + j] = a1[j]; }
#pragma unroll
    for (int g = 0; g < 4; ++g) {
      const bf16_t* bp =
          Bp + ((size_t)(ntbase + g) * ktiles + kt) * 512 + lane * 16;
      v16bf b = *(const v16bf*)bp;
      acc[g] = __builtin_amdgcn_wmma_f32_16x16x32_bf16(
          false, a, false, b, (short)0, acc[g], false, false);
    }
  }

  // Epilogue. C layout: VGPR r -> M = r (+8 for lanes 16..31), N = lane&15.
  const int mbase = mt * 16 + (lane >> 4) * 8;
#pragma unroll
  for (int g = 0; g < 4; ++g) {
    int n = (ntbase + g) * 16 + (lane & 15);
    float bv = bias ? bias[n] : 0.0f;
#pragma unroll
    for (int r = 0; r < 8; ++r) {
      int m = mbase + r;
      float v = acc[g][r] + bv;
      if (Cf)  Cf[(size_t)m * N + n] = v;
      if (Cbf) Cbf[(size_t)m * N + n] = (bf16_t)v;
    }
  }
}

// --------------------------------------------------------------------------
// Recurrent scan: one workgroup (32 waves) per direction, sequential over T.
// h (16 x 512 bf16, batch padded 8->16 with zero rows) and c (8 x 512 f32)
// live in LDS. Wave w owns hidden columns [16w, 16w+16); its 4 WMMA
// accumulators are the i/f/g/o gates of those SAME columns, so the cell
// update is wave-local. Wh streams from L2 (2 MB/step) in packed fragments.
// xp gate preactivations for step t are prefetched BEFORE the WMMA phase
// (no dependence on h) so their latency hides under the matrix work.
// Backward direction scatters h_t to the un-flipped time slot directly.
// --------------------------------------------------------------------------
__global__ __launch_bounds__(1024) void lstm_recurrent_kernel(
    const float* __restrict__ xp_f, const float* __restrict__ xp_b,
    const bf16_t* __restrict__ Whf_p, const bf16_t* __restrict__ Whb_p,
    const int* __restrict__ seq, bf16_t* __restrict__ ybf) {
  const int dir = blockIdx.x;
  const float* xp   = dir ? xp_b : xp_f;
  const bf16_t* Whp = dir ? Whb_p : Whf_p;

  __shared__ __align__(16) bf16_t h_sh[16 * Hn];  // 16 KB
  __shared__ __align__(16) float  c_sh[Bn * Hn];  // 16 KB
  __shared__ int Lsh[Bn];

  const int tid  = threadIdx.x;
  const int lane = tid & 31;
  const int wv   = tid >> 5;  // 0..31 : hidden column group

  for (int i = tid; i < 16 * Hn; i += 1024) h_sh[i] = (bf16_t)0.0f;
  for (int i = tid; i < Bn * Hn; i += 1024) c_sh[i] = 0.0f;
  if (tid < Bn) Lsh[tid] = seq[tid];
  __syncthreads();

  const int arow  = lane & 15;
  const int khalf = (lane >> 4) * 16;
  const int col16 = wv * 16;
  const int col   = col16 + (lane & 15);

  for (int t = 0; t < Tn; ++t) {
    // Prefetch this step's gate preactivations (overlaps the WMMA phase).
    float xpre[4][Bn];
    if (lane < 16) {
#pragma unroll
      for (int r = 0; r < Bn; ++r) {
        size_t xo = ((size_t)r * Tn + t) * G4 + col;
        xpre[0][r] = xp[xo];
        xpre[1][r] = xp[xo + 512];
        xpre[2][r] = xp[xo + 1024];
        xpre[3][r] = xp[xo + 1536];
      }
    }

    v8f acc[4];
    for (int g = 0; g < 4; ++g)
      for (int r = 0; r < 8; ++r) acc[g][r] = 0.0f;

#pragma unroll 4
    for (int kt = 0; kt < 16; ++kt) {
      const bf16_t* ap = &h_sh[arow * Hn + kt * 32 + khalf];
      v8bf a0 = *(const v8bf*)ap;
      v8bf a1 = *(const v8bf*)(ap + 8);
      v16bf a;
      for (int j = 0; j < 8; ++j) { a[j] = a0[j]; a[8 + j] = a1[j]; }
#pragma unroll
      for (int g = 0; g < 4; ++g) {
        int nt = g * 32 + wv;  // gate g, columns col16..col16+15
        const bf16_t* bp = Whp + ((size_t)nt * 16 + kt) * 512 + lane * 16;
        v16bf b = *(const v16bf*)bp;
        acc[g] = __builtin_amdgcn_wmma_f32_16x16x32_bf16(
            false, a, false, b, (short)0, acc[g], false, false);
      }
    }
    __syncthreads();  // everyone finished reading h_sh for step t

    if (lane < 16) {  // lanes 0..15 hold batch rows M = r = 0..7 (valid)
#pragma unroll
      for (int r = 0; r < Bn; ++r) {
        float iv = acc[0][r] + xpre[0][r];
        float fv = acc[1][r] + xpre[1][r];
        float gv = acc[2][r] + xpre[2][r];
        float ov = acc[3][r] + xpre[3][r];
        float cold = c_sh[r * Hn + col];
        float cn = sigmoidf_(fv) * cold + sigmoidf_(iv) * tanhf(gv);
        float hn = sigmoidf_(ov) * tanhf(cn);
        c_sh[r * Hn + col] = cn;
        h_sh[r * Hn + col] = (bf16_t)hn;
        int L = Lsh[r];
        int tout = dir ? ((t < L) ? (L - 1 - t) : t) : t;
        ybf[((size_t)r * Tn + tout) * H2 + dir * Hn + col] = (bf16_t)hn;
      }
    }
    __syncthreads();  // h_sh updated before step t+1 reads it
  }
}

// --------------------------------------------------------------------------
// Host orchestration.
// Workspace layout (~355 MB): packed bf16 weights, bf16 activations,
// f32 gate preactivations.
// --------------------------------------------------------------------------
extern "C" void kernel_launch(void* const* d_in, const int* in_sizes, int n_in,
                              void* d_out, int out_size, void* d_ws,
                              size_t ws_size, hipStream_t stream) {
  const float* inputs = (const float*)d_in[0];
  const int*   seq    = (const int*)d_in[1];
  const float* Wx_f   = (const float*)d_in[2];
  const float* Wh_f   = (const float*)d_in[3];
  const float* b_f    = (const float*)d_in[4];
  const float* Wx_b   = (const float*)d_in[5];
  const float* Wh_b   = (const float*)d_in[6];
  const float* b_b    = (const float*)d_in[7];
  const float* Wp     = (const float*)d_in[8];
  const float* bp     = (const float*)d_in[9];
  float* out = (float*)d_out;

  char* ws = (char*)d_ws;
  size_t off = 0;
  auto take = [&](size_t bytes) -> char* {
    char* p = ws + off;
    off += (bytes + 255) & ~(size_t)255;
    return p;
  };

  const size_t WXe = (size_t)Dn * G4;
  const size_t WHe = (size_t)Hn * G4;
  const size_t WPe = (size_t)H2 * Dn;

  bf16_t *wxfp[NLAYERS], *whfp[NLAYERS], *wxbp[NLAYERS], *whbp[NLAYERS],
         *wpp[NLAYERS];
  for (int l = 0; l < NLAYERS; ++l) {
    wxfp[l] = (bf16_t*)take(WXe * 2);
    whfp[l] = (bf16_t*)take(WHe * 2);
    wxbp[l] = (bf16_t*)take(WXe * 2);
    whbp[l] = (bf16_t*)take(WHe * 2);
    wpp[l]  = (bf16_t*)take(WPe * 2);
  }
  bf16_t* xbf  = (bf16_t*)take((size_t)MT * Dn * 2);  // layer input, bf16
  bf16_t* xrbf = (bf16_t*)take((size_t)MT * Dn * 2);  // flipped input
  bf16_t* ybf  = (bf16_t*)take((size_t)MT * H2 * 2);  // [yf|yb] concat
  float*  xpf  = (float*)take((size_t)MT * G4 * 4);   // fwd preactivations
  float*  xpb  = (float*)take((size_t)MT * G4 * 4);   // bwd preactivations

  // ---- weight repack (once per launch; deterministic) ----
  for (int l = 0; l < NLAYERS; ++l) {
    repack_b_kernel<<<(int)((WXe + 255) / 256), 256, 0, stream>>>(
        Wx_f + l * WXe, wxfp[l], Dn, G4);
    repack_b_kernel<<<(int)((WHe + 255) / 256), 256, 0, stream>>>(
        Wh_f + l * WHe, whfp[l], Hn, G4);
    repack_b_kernel<<<(int)((WXe + 255) / 256), 256, 0, stream>>>(
        Wx_b + l * WXe, wxbp[l], Dn, G4);
    repack_b_kernel<<<(int)((WHe + 255) / 256), 256, 0, stream>>>(
        Wh_b + l * WHe, whbp[l], Hn, G4);
    repack_b_kernel<<<(int)((WPe + 255) / 256), 256, 0, stream>>>(
        Wp + l * WPe, wpp[l], H2, Dn);
  }

  const int nx = MT * Dn;
  f32_to_bf16_kernel<<<(nx + 255) / 256, 256, 0, stream>>>(inputs, xbf, nx);

  for (int l = 0; l < NLAYERS; ++l) {
    // flipped input for the backward LSTM
    flip_bf16_kernel<<<(nx + 255) / 256, 256, 0, stream>>>(xbf, xrbf, seq, nx);

    // xp = x @ Wx + b  (both directions). Grid: (M/16) x (N/512) blocks.
    int xpBlocks = (MT / 16) * (G4 / 512);
    wmma_gemm_kernel<<<xpBlocks, 256, 0, stream>>>(
        xbf, wxfp[l], b_f + l * G4, xpf, nullptr, MT, G4, Dn);
    wmma_gemm_kernel<<<xpBlocks, 256, 0, stream>>>(
        xrbf, wxbp[l], b_b + l * G4, xpb, nullptr, MT, G4, Dn);

    // sequential scan, fwd + bwd concurrently (one WGP each)
    lstm_recurrent_kernel<<<2, 1024, 0, stream>>>(xpf, xpb, whfp[l], whbp[l],
                                                  seq, ybf);

    // projection: x_next = [yf|yb] @ Wp + bp (f32 to out, bf16 to xbf)
    int pBlocks = (MT / 16) * (Dn / 512);
    wmma_gemm_kernel<<<pBlocks, 256, 0, stream>>>(
        ybf, wpp[l], bp + l * Dn, out, xbf, MT, Dn, H2);
  }
  (void)in_sizes; (void)n_in; (void)out_size; (void)ws_size;
}
